// EdgeNetwork_26182120636655
// MI455X (gfx1250) — matrix operand order repaired and checked by reading, not compiled, not run on hardware
//
#include <hip/hip_runtime.h>

typedef __attribute__((ext_vector_type(16))) _Float16 v16h;
typedef __attribute__((ext_vector_type(8)))  float    v8f;
typedef __attribute__((ext_vector_type(4)))  float    v4f;

#define INPUT_DIM 16
#define HIDDEN    8

// ---- fast transcendentals on the TRANS32 pipe ----
__device__ __forceinline__ float fast_exp(float x) {
    return __builtin_amdgcn_exp2f(x * 1.44269504088896f);   // v_exp_f32
}
__device__ __forceinline__ float fast_rcp(float x) {
    return __builtin_amdgcn_rcpf(x);                        // v_rcp_f32
}
__device__ __forceinline__ float fast_tanh(float x) {
    return 1.0f - 2.0f * fast_rcp(fast_exp(2.0f * x) + 1.0f);
}
__device__ __forceinline__ float fast_sigmoid(float x) {
    return fast_rcp(1.0f + fast_exp(-x));
}

// ---- DPP8 xor-add reduction within 8-lane rows (VALU only, no LDS path) ----
#define DPP8_SEL(a,b,c,d,e,f,g,h) \
    ((a)|((b)<<3)|((c)<<6)|((d)<<9)|((e)<<12)|((f)<<15)|((g)<<18)|((h)<<21))

template<int SEL>
__device__ __forceinline__ float dpp8_xor_add(float v) {
    int i = __builtin_bit_cast(int, v);
    int j = __builtin_amdgcn_mov_dpp8(i, SEL);
    return v + __builtin_bit_cast(float, j);
}
__device__ __forceinline__ float row8_reduce(float v) {
    v = dpp8_xor_add<DPP8_SEL(1,0,3,2,5,4,7,6)>(v);  // xor 1
    v = dpp8_xor_add<DPP8_SEL(2,3,0,1,6,7,4,5)>(v);  // xor 2
    v = dpp8_xor_add<DPP8_SEL(4,5,6,7,0,1,2,3)>(v);  // xor 4
    return v;
}

// ---- gather one A-tile row-pair for this lane and pack to f16 ----
// A layout (16-bit 16x32): lane holds row M=lane&15; elems 0..7 = K kbase..kbase+7
// (x[col]); elems 8..15 = K 16+kbase.. (x[row]); kbase = (lane>=16)?8:0.
__device__ __forceinline__ v16h load_a_tile(const float* __restrict__ x,
                                            const long long* __restrict__ rowp,
                                            const long long* __restrict__ colp,
                                            int eld, int kbase) {
    const int ci = (int)colp[eld];
    const int ri = (int)rowp[eld];
    const float* xc = x + (size_t)ci * INPUT_DIM + kbase;
    const float* xr = x + (size_t)ri * INPUT_DIM + kbase;
    const v4f c0 = *(const v4f*)(xc);
    const v4f c1 = *(const v4f*)(xc + 4);
    const v4f r0 = *(const v4f*)(xr);
    const v4f r1 = *(const v4f*)(xr + 4);
    v16h a;
    #pragma unroll
    for (int j = 0; j < 4; ++j) {
        a[j]      = (_Float16)c0[j];
        a[j + 4]  = (_Float16)c1[j];
        a[j + 8]  = (_Float16)r0[j];
        a[j + 12] = (_Float16)r1[j];
    }
    return a;
}

// ---- second layer + activations + store for one 16-edge tile ----
__device__ __forceinline__ void epilogue(v8f acc, int n, int hi, int eb, int E,
                                         float wn, float bias2,
                                         float* __restrict__ out) {
    float mine = 0.0f;
    #pragma unroll
    for (int r = 0; r < 8; ++r) {
        float v = fast_tanh(acc[r]) * wn;   // bias b1 already in acc; wn==0 for n>=8
        v = row8_reduce(v);                 // sum over hidden units (lanes 0..7 of row)
        if (r == n) mine = v;               // lane n keeps the sum for edge eb+n+8*hi
    }
    if (n < 8) {
        const int eo = eb + n + 8 * hi;
        if (eo < E) out[eo] = fast_sigmoid(mine + bias2);
    }
}

__global__ __launch_bounds__(256)
void edge_mlp_wmma(const float* __restrict__ x,
                   const long long* __restrict__ edge_index,
                   const float* __restrict__ W1,
                   const float* __restrict__ b1,
                   const float* __restrict__ W2,
                   const float* __restrict__ b2,
                   float* __restrict__ out,
                   int E)
{
    const int lane  = threadIdx.x & 31;
    const int wid   = blockIdx.x * (blockDim.x >> 5) + (threadIdx.x >> 5);
    const int nw    = gridDim.x * (blockDim.x >> 5);

    const int n     = lane & 15;      // C/D column (hidden index) & edge-in-tile id
    const int hi    = lane >> 4;
    const int kbase = hi * 8;

    // B matrix: lane holds column n of W1[32x8] (cols 8..15 zero-padded); elems 0..7 = K kbase.., 8..15 = K 16+kbase..
    v16h bmat;
    {
        #pragma unroll
        for (int j = 0; j < 16; ++j) {
            const int k = (j < 8) ? (kbase + j) : (16 + kbase + j - 8);
            bmat[j] = (n < HIDDEN) ? (_Float16)W1[k * HIDDEN + n] : (_Float16)0.0f;
        }
    }

    const float bn    = (n < HIDDEN) ? b1[n] : 0.0f;
    const float wn    = (n < HIDDEN) ? W2[n] : 0.0f;
    const float bias2 = b2[0];

    // bias b1 pre-loaded into the WMMA accumulator (C input)
    v8f accb;
    #pragma unroll
    for (int i = 0; i < 8; ++i) accb[i] = bn;

    const long long* rowp = edge_index;        // features 16..31 come from x[row]
    const long long* colp = edge_index + E;    // features  0..15 come from x[col]

    const int ntiles = (E + 15) >> 4;

    // two tiles per iteration: second WMMA + epilogues fill the WMMA->VALU hazard
    int t = wid * 2;
    for (; t + 1 < ntiles; t += nw * 2) {
        const int eb0 = t << 4;
        const int eb1 = (t + 1) << 4;
        int e0 = eb0 + n;  if (e0 >= E) e0 = E - 1;   // clamp: EXEC stays full
        int e1 = eb1 + n;  if (e1 >= E) e1 = E - 1;

        // prefetch next iteration's index cachelines (global_prefetch_b8;
        // no LOADcnt cost) -- shortens the idx->gather->wmma dependent chain
        {
            const int tn = t + nw * 2;
            if (tn + 1 < ntiles) {
                const int en = (tn << 4) + n;
                __builtin_prefetch(&colp[en], 0, 3);
                __builtin_prefetch(&rowp[en], 0, 3);
            }
        }

        const v16h a0 = load_a_tile(x, rowp, colp, e0, kbase);
        const v16h a1 = load_a_tile(x, rowp, colp, e1, kbase);

        v8f acc0 = __builtin_amdgcn_wmma_f32_16x16x32_f16(
                       false, a0, false, bmat, (short)0, accb, false, false);
        v8f acc1 = __builtin_amdgcn_wmma_f32_16x16x32_f16(
                       false, a1, false, bmat, (short)0, accb, false, false);

        epilogue(acc0, n, hi, eb0, E, wn, bias2, out);
        epilogue(acc1, n, hi, eb1, E, wn, bias2, out);
    }
    // tail: at most one leftover tile for this wave
    if (t < ntiles) {
        const int eb = t << 4;
        int e = eb + n;  if (e >= E) e = E - 1;
        const v16h a = load_a_tile(x, rowp, colp, e, kbase);
        v8f acc = __builtin_amdgcn_wmma_f32_16x16x32_f16(
                      false, a, false, bmat, (short)0, accb, false, false);
        epilogue(acc, n, hi, eb, E, wn, bias2, out);
    }
}

extern "C" void kernel_launch(void* const* d_in, const int* in_sizes, int n_in,
                              void* d_out, int out_size, void* d_ws, size_t ws_size,
                              hipStream_t stream) {
    const float*     x   = (const float*)d_in[0];
    const long long* ei  = (const long long*)d_in[1];   // int64 per reference
    const float*     W1  = (const float*)d_in[2];
    const float*     b1  = (const float*)d_in[3];
    const float*     W2  = (const float*)d_in[4];
    const float*     b2  = (const float*)d_in[5];
    float*           out = (float*)d_out;

    const int E = in_sizes[1] / 2;            // edge_index is [2, E]
    const int ntiles = (E + 15) / 16;
    const int wavesPerBlock = 256 / 32;
    const int pairs = (ntiles + 1) / 2;
    int blocks = (pairs + wavesPerBlock - 1) / wavesPerBlock;
    if (blocks > 4096) blocks = 4096;         // grid-stride loop covers the rest
    if (blocks < 1) blocks = 1;

    edge_mlp_wmma<<<blocks, 256, 0, stream>>>(x, ei, W1, b1, W2, b2, out, E);
}
